// CausalSelfAttention_67757404062002
// MI455X (gfx1250) — matrix operand order repaired and checked by reading, not compiled, not run on hardware
//
#include <hip/hip_runtime.h>
#include <hip/hip_bf16.h>

// ---------------------------------------------------------------------------
// Causal self-attention (B=2, T=2048, C=1024, H=16, HS=64) for gfx1250.
// Matmuls on v_wmma_f32_16x16x32_f16; tile staging via
// global_load_async_to_lds_b128 (ASYNCcnt) + s_wait_asynccnt.
// GEMM: 64x64 wave tiles (16 WMMA / K-step) to maximize math per LDS read.
// ---------------------------------------------------------------------------

typedef __attribute__((ext_vector_type(16))) _Float16 v16h;
typedef __attribute__((ext_vector_type(8)))  float    v8f;

constexpr int Bb = 2;
constexpr int Tt = 2048;
constexpr int Cc = 1024;
constexpr int Hh = 16;
constexpr int HS = 64;         // head size
constexpr int C3 = 3 * Cc;     // 3072
constexpr int Mrows = Bb * Tt; // 4096

// ---------------------------------------------------------------------------
// CDNA5 helpers
// ---------------------------------------------------------------------------
__device__ __forceinline__ v8f wmma16(v16h a, v16h b, v8f c) {
  return __builtin_amdgcn_wmma_f32_16x16x32_f16(false, a, false, b, (short)0, c,
                                                false, false);
}

// Async copy 16 bytes from global to LDS (per active lane). Tracked by ASYNCcnt.
__device__ __forceinline__ void async_cp16(_Float16* lds_ptr, const _Float16* gptr) {
  unsigned lds_off = (unsigned)(size_t)lds_ptr;            // addr[31:0] = LDS offset
  unsigned long long ga = (unsigned long long)(size_t)gptr;
  asm volatile("global_load_async_to_lds_b128 %0, %1, off"
               :: "v"(lds_off), "v"(ga) : "memory");
}

__device__ __forceinline__ void wait_async0() {
  asm volatile("s_wait_asynccnt 0" ::: "memory");
}

// A-operand fragment: 16(M) x 32(K) f16 tile, row-major, row stride `ld` halves.
// Lanes 0-15 -> M=lane, K in {0..7,16..23}; lanes 16-31 -> K in {8..15,24..31}.
__device__ __forceinline__ v16h load_frag_a(const _Float16* base, int ld) {
  const int lane = threadIdx.x & 31;
  const int m = lane & 15;
  const int kh = (lane >> 4) << 3;
  const _Float16* p = base + m * ld + kh;
  union { v16h v; unsigned u[8]; } r;
#pragma unroll
  for (int i = 0; i < 4; ++i) {
    r.u[i]     = *(const unsigned*)(p + 2 * i);
    r.u[i + 4] = *(const unsigned*)(p + 16 + 2 * i);
  }
  return r.v;
}

// B-operand fragment: 32(K) x 16(N), stored TRANSPOSED as [n][k] row-major.
// Lanes 0-15 -> N=lane, K in {0..7,16..23}; lanes 16-31 -> K in {8..15,24..31}.
__device__ __forceinline__ v16h load_frag_b(const _Float16* base, int ld) {
  const int lane = threadIdx.x & 31;
  const int n = lane & 15;
  const int kh = (lane >> 4) << 3;
  const _Float16* p = base + n * ld + kh;
  union { v16h v; unsigned u[8]; } r;
#pragma unroll
  for (int i = 0; i < 4; ++i) {
    r.u[i]     = *(const unsigned*)(p + 2 * i);
    r.u[i + 4] = *(const unsigned*)(p + 16 + 2 * i);
  }
  return r.v;
}

// ---------------------------------------------------------------------------
// Cast kernels
// ---------------------------------------------------------------------------
__global__ __launch_bounds__(256) void cast_f16_kernel(const float* __restrict__ in,
                                                       _Float16* __restrict__ out, int n) {
  int i = blockIdx.x * blockDim.x + threadIdx.x;
  if (i < n) out[i] = (_Float16)in[i];
}

__global__ __launch_bounds__(256) void cast_transpose_kernel(const float* __restrict__ in,
                                                             _Float16* __restrict__ out,
                                                             int rows, int cols) {
  int i = blockIdx.x * blockDim.x + threadIdx.x;
  if (i < rows * cols) {
    int r = i / cols, c = i % cols;
    out[(size_t)c * rows + r] = (_Float16)in[i];
  }
}

// ---------------------------------------------------------------------------
// WMMA GEMM: out[M,N] = A[M,K] * Bt[N,K]^T + bias[N]
// 128 threads (4 waves), block tile 128x128, K-step 32, double-buffered LDS
// filled by async-to-LDS DMA.  Wave w (2x2 grid): 64x64 tile -> 4x4 WMMA.
// ---------------------------------------------------------------------------
constexpr int BM = 128, BN = 128, BK = 32;
constexpr int SA = 40;   // padded LDS row stride (halves)

__device__ __forceinline__ void load_tiles_async(const _Float16* __restrict__ A,
                                                 const _Float16* __restrict__ Bt,
                                                 _Float16* as, _Float16* bs,
                                                 int bm, int bn, int K, int k0, int tid) {
  // 512 x 16B chunks per operand tile; 128 threads -> 4 chunks each per operand
#pragma unroll
  for (int it = 0; it < 4; ++it) {
    int i = tid + it * 128;
    int row = i >> 2;               // 0..127
    int seg = (i & 3) << 3;         // 0,8,16,24 halves
    async_cp16(&as[row * SA + seg], A + (size_t)(bm + row) * K + k0 + seg);
    async_cp16(&bs[row * SA + seg], Bt + (size_t)(bn + row) * K + k0 + seg);
  }
}

template <bool OUT_F16>
__global__ __launch_bounds__(128) void wmma_gemm_kernel(const _Float16* __restrict__ A,
                                                        const _Float16* __restrict__ Bt,
                                                        const float* __restrict__ bias,
                                                        void* __restrict__ out,
                                                        int M, int N, int K) {
  __shared__ _Float16 As[2][BM * SA];
  __shared__ _Float16 Bs[2][BN * SA];

  const int tid  = threadIdx.x;
  const int wave = tid >> 5;           // 0..3
  const int wm   = (wave & 1) * 64;
  const int wn   = (wave >> 1) * 64;
  const int bm   = blockIdx.x * BM;
  const int bn   = blockIdx.y * BN;

  v8f acc[4][4] = {};

  load_tiles_async(A, Bt, As[0], Bs[0], bm, bn, K, 0, tid);
  const int nk = K / BK;
  for (int kk = 0; kk < nk; ++kk) {
    const int buf = kk & 1;
    wait_async0();        // own async fills of `buf` complete
    __syncthreads();      // everyone's fills visible; prev buffer fully consumed
    if (kk + 1 < nk)
      load_tiles_async(A, Bt, As[buf ^ 1], Bs[buf ^ 1], bm, bn, K, (kk + 1) * BK, tid);
    if (kk + 2 < nk) {    // pull tile kk+2 toward L2 while DMA fills kk+1
      const int prow = tid & 127;
      __builtin_prefetch(A + (size_t)(bm + prow) * K + (kk + 2) * BK, 0, 1);
      __builtin_prefetch(Bt + (size_t)(bn + prow) * K + (kk + 2) * BK, 0, 1);
    }

    v16h af[4], bf[4];
#pragma unroll
    for (int i = 0; i < 4; ++i) af[i] = load_frag_a(&As[buf][(wm + i * 16) * SA], SA);
#pragma unroll
    for (int j = 0; j < 4; ++j) bf[j] = load_frag_b(&Bs[buf][(wn + j * 16) * SA], SA);
#pragma unroll
    for (int i = 0; i < 4; ++i)
#pragma unroll
      for (int j = 0; j < 4; ++j)
        acc[i][j] = wmma16(af[i], bf[j], acc[i][j]);
  }

  const int ncol = tid & 15;
  const int mb   = ((tid & 31) >> 4) << 3;
#pragma unroll
  for (int i = 0; i < 4; ++i) {
#pragma unroll
    for (int j = 0; j < 4; ++j) {
      const int gn = bn + wn + j * 16 + ncol;
      const float bv = bias[gn];
      const int gm0 = bm + wm + i * 16 + mb;
#pragma unroll
      for (int r = 0; r < 8; ++r) {
        float v = acc[i][j][r] + bv;
        if (OUT_F16)
          ((_Float16*)out)[(size_t)(gm0 + r) * N + gn] = (_Float16)v;
        else
          ((float*)out)[(size_t)(gm0 + r) * N + gn] = v;
      }
    }
  }
}

// ---------------------------------------------------------------------------
// Flash attention (causal), per (b,h).  qkv: f16 [B*T, 3C] (q|k|v cols).
// 128 threads (4 waves); block = 64 query rows, wave = 16.  Key tiles of 32.
// K tile staged by async-to-LDS; V staged transposed via scalar path.
// ---------------------------------------------------------------------------
constexpr int LDK = 72;  // Ks row stride: [key][d]
constexpr int LDV = 40;  // Vs row stride: [d][key]

__global__ __launch_bounds__(128) void flash_attn_kernel(const _Float16* __restrict__ qkv,
                                                         _Float16* __restrict__ attn_out) {
  __shared__ _Float16 Ks[32 * LDK];
  __shared__ _Float16 Vs[64 * LDV];
  __shared__ _Float16 Ps[4][16 * 32];

  const int tid  = threadIdx.x;
  const int wave = tid >> 5;
  const int lane = tid & 31;
  const int ncol = lane & 15;
  const int mb   = (lane >> 4) << 3;

  const int b = blockIdx.y / Hh;
  const int h = blockIdx.y % Hh;
  const int q0 = blockIdx.x * 64;
  const int qrow = q0 + wave * 16;

  const size_t rowbase = (size_t)b * Tt * C3;
  const _Float16* Qp = qkv + rowbase + (size_t)qrow * C3 + h * HS;
  const _Float16* Kp = qkv + rowbase + Cc + h * HS;
  const _Float16* Vp = qkv + rowbase + 2 * Cc + h * HS;

  const v16h qa0 = load_frag_a(Qp, C3);
  const v16h qa1 = load_frag_a(Qp + 32, C3);

  v8f o0 = {}, o1 = {}, o2 = {}, o3 = {};
  float m_run[8], l_run[8];
#pragma unroll
  for (int r = 0; r < 8; ++r) { m_run[r] = -1e30f; l_run[r] = 0.f; }

  const float scale = 0.125f;     // HS^-0.5
  const int nkt = (q0 >> 5) + 2;  // covers keys <= q0+63

  for (int kt = 0; kt < nkt; ++kt) {
    const int kbase = kt * 32;
    __syncthreads();              // prev tile fully consumed by all waves
    // K tile -> Ks[key][d] via async DMA (32x64 halves = 256 x 16B)
#pragma unroll
    for (int it = 0; it < 2; ++it) {
      int i = tid + it * 128;
      int key = i >> 3, seg = (i & 7) << 3;
      async_cp16(&Ks[key * LDK + seg], Kp + (size_t)(kbase + key) * C3 + seg);
    }
    // V tile transposed -> Vs[d][key]
#pragma unroll
    for (int it = 0; it < 16; ++it) {
      int i = tid + it * 128;
      int key = i >> 6, d = i & 63;
      Vs[d * LDV + key] = Vp[(size_t)(kbase + key) * C3 + d];
    }
    wait_async0();
    __syncthreads();

    // scores: s0 = keys [0,16), s1 = keys [16,32)
    v8f s0 = {}, s1 = {};
    s0 = wmma16(qa0, load_frag_b(&Ks[0], LDK), s0);
    s0 = wmma16(qa1, load_frag_b(&Ks[32], LDK), s0);
    s1 = wmma16(qa0, load_frag_b(&Ks[16 * LDK], LDK), s1);
    s1 = wmma16(qa1, load_frag_b(&Ks[16 * LDK + 32], LDK), s1);

    float tmax[8];
#pragma unroll
    for (int r = 0; r < 8; ++r) {
      const int qr = qrow + mb + r;
      float v0 = s0[r] * scale;
      float v1 = s1[r] * scale;
      if (kbase + ncol > qr)      v0 = -1e30f;
      if (kbase + 16 + ncol > qr) v1 = -1e30f;
      s0[r] = v0; s1[r] = v1;
      tmax[r] = fmaxf(v0, v1);
    }
#pragma unroll
    for (int s = 8; s >= 1; s >>= 1)
#pragma unroll
      for (int r = 0; r < 8; ++r)
        tmax[r] = fmaxf(tmax[r], __shfl_xor(tmax[r], s, 32));

    float rsum[8];
#pragma unroll
    for (int r = 0; r < 8; ++r) {
      const float mnew  = fmaxf(m_run[r], tmax[r]);
      const float alpha = __expf(m_run[r] - mnew);
      m_run[r] = mnew;
      const float p0 = __expf(s0[r] - mnew);
      const float p1 = __expf(s1[r] - mnew);
      s0[r] = p0; s1[r] = p1;
      rsum[r] = p0 + p1;
      o0[r] *= alpha; o1[r] *= alpha; o2[r] *= alpha; o3[r] *= alpha;
      l_run[r] *= alpha;
    }
#pragma unroll
    for (int s = 8; s >= 1; s >>= 1)
#pragma unroll
      for (int r = 0; r < 8; ++r)
        rsum[r] += __shfl_xor(rsum[r], s, 32);
#pragma unroll
    for (int r = 0; r < 8; ++r) l_run[r] += rsum[r];

    // P: C-layout -> per-wave LDS scratch -> A fragment (same-wave DS in order)
    _Float16* Pw = &Ps[wave][0];
#pragma unroll
    for (int r = 0; r < 8; ++r) {
      Pw[(mb + r) * 32 + ncol]      = (_Float16)s0[r];
      Pw[(mb + r) * 32 + 16 + ncol] = (_Float16)s1[r];
    }
    const v16h pa = load_frag_a(Pw, 32);

    o0 = wmma16(pa, load_frag_b(&Vs[0 * LDV], LDV), o0);
    o1 = wmma16(pa, load_frag_b(&Vs[16 * LDV], LDV), o1);
    o2 = wmma16(pa, load_frag_b(&Vs[32 * LDV], LDV), o2);
    o3 = wmma16(pa, load_frag_b(&Vs[48 * LDV], LDV), o3);
  }

#pragma unroll
  for (int r = 0; r < 8; ++r) {
    const float invl = 1.f / l_run[r];
    _Float16* Orow = attn_out + ((size_t)(b * Tt) + qrow + mb + r) * Cc + h * HS;
    Orow[ncol]      = (_Float16)(o0[r] * invl);
    Orow[16 + ncol] = (_Float16)(o1[r] * invl);
    Orow[32 + ncol] = (_Float16)(o2[r] * invl);
    Orow[48 + ncol] = (_Float16)(o3[r] * invl);
  }
}

// ---------------------------------------------------------------------------
// Launcher
// ---------------------------------------------------------------------------
extern "C" void kernel_launch(void* const* d_in, const int* in_sizes, int n_in,
                              void* d_out, int out_size, void* d_ws, size_t ws_size,
                              hipStream_t stream) {
  const float* x      = (const float*)d_in[0];
  const float* W_attn = (const float*)d_in[1];
  const float* b_attn = (const float*)d_in[2];
  const float* W_proj = (const float*)d_in[3];
  const float* b_proj = (const float*)d_in[4];
  float* out = (float*)d_out;

  // Workspace (f16): x16(8M) | Wat(6M) | Wpt(2M) | qkv(24M) | attn(8M)
  char* ws = (char*)d_ws;
  _Float16* x16    = (_Float16*)(ws);
  _Float16* Wat    = (_Float16*)(ws + ((size_t)8  << 20));
  _Float16* Wpt    = (_Float16*)(ws + ((size_t)14 << 20));
  _Float16* qkv16  = (_Float16*)(ws + ((size_t)16 << 20));
  _Float16* attn16 = (_Float16*)(ws + ((size_t)40 << 20));

  const int nx = Mrows * Cc;
  cast_f16_kernel<<<(nx + 255) / 256, 256, 0, stream>>>(x, x16, nx);
  cast_transpose_kernel<<<(Cc * C3 + 255) / 256, 256, 0, stream>>>(W_attn, Wat, Cc, C3);
  cast_transpose_kernel<<<(Cc * Cc + 255) / 256, 256, 0, stream>>>(W_proj, Wpt, Cc, Cc);

  wmma_gemm_kernel<true><<<dim3(Mrows / BM, C3 / BN), 128, 0, stream>>>(
      x16, Wat, b_attn, qkv16, Mrows, C3, Cc);

  flash_attn_kernel<<<dim3(Tt / 64, Bb * Hh), 128, 0, stream>>>(qkv16, attn16);

  wmma_gemm_kernel<false><<<dim3(Mrows / BM, Cc / BN), 128, 0, stream>>>(
      attn16, Wpt, b_proj, out, Mrows, Cc, Cc);
}
